// TGCN2_3693671874750
// MI455X (gfx1250) — matrix operand order repaired and checked by reading, not compiled, not run on hardware
//
#include <hip/hip_runtime.h>

// ---------------------------------------------------------------------------
// TGCN (gated triplet GCN) for MI455X / gfx1250, bf16 WMMA path + TDM staging.
//
// Reference simplifications (exact):
//   * Hn0/He0 are zeros -> 'r' gate is dead code, gate linears use W[:, :H],
//     and H_out = (1 - sigmoid_z) * tanh_h.
//
// Input flattening assumption (recursive dict-insertion order of setup_inputs):
//   d_in[0]=X, d_in[1]=edge_index, d_in[2]=edge_feature,
//   gates z(base 3), r(base 27, unused), h(base 51); per-gate leaf order:
//   +0 nn1.lin1.W +1 b  +2 ln1.g +3 ln1.b  +4 nn1.lin2.W +5 b  +6 ln2.g +7 ln2.b
//   +8 nn2.lin1.W +9 b  +10 nn2.ln1.g +11 b  +12 nn2.lin2.W +13 b
//   +14 nn3.lin1.W +15 b +16 nn3.ln1.g +17 b +18 nn3.lin2.W +19 b
//   +20 node.W +21 node.b +22 edge.W +23 edge.b
// ---------------------------------------------------------------------------

typedef __bf16 v16bf  __attribute__((ext_vector_type(16)));
typedef __bf16 bf16x8 __attribute__((ext_vector_type(8)));
typedef float  v8f    __attribute__((ext_vector_type(8)));
typedef unsigned int u32x4 __attribute__((ext_vector_type(4)));
typedef int          i32x4 __attribute__((ext_vector_type(4)));
typedef int          i32x8 __attribute__((ext_vector_type(8)));

#define N_EDGES 80000
#define N_NODES 10000
#define N_BATCH 2

#if defined(__has_builtin)
#if __has_builtin(__builtin_amdgcn_tensor_load_to_lds) && \
    __has_builtin(__builtin_amdgcn_s_wait_tensorcnt)
#define HAVE_TDM 1
#endif
#endif

__device__ __forceinline__ float wave_sum(float v) {
#pragma unroll
  for (int o = 16; o > 0; o >>= 1) v += __shfl_xor(v, o, 32);
  return v;
}

// AMODE: 0 = gather concat [x_tgt|ef|x_src] (K=384), 1 = bf16 ws rows (TDM), 2 = f32 ws rows
// EMODE: 0 = bias+LN+ReLU -> bf16 out
//        1 = bias         -> bf16 out
//        2 = bias+LN+ReLU -> msg scatter-atomicAdd(agg) + new_e bf16 (N=640)
//        3 = sigmoid(bias+x)            -> f32 out (d_out region)
//        4 = (1 - out) * tanh(bias + x) -> f32 out in place (reads sigmoid_z)
template <int AMODE, int EMODE, int KDIM, int NDIM, int NW>
__global__ void tgcn_gemm(const float* __restrict__ X,
                          const int* __restrict__ EI,
                          const float* __restrict__ EF,
                          const __bf16* __restrict__ Abf,
                          const float* __restrict__ Af32,
                          const __bf16* __restrict__ W,     // N x K row-major bf16
                          const float* __restrict__ bias,
                          const float* __restrict__ lng,
                          const float* __restrict__ lnb,
                          __bf16* __restrict__ outbf,
                          float* __restrict__ agg,
                          __bf16* __restrict__ newe,
                          float* __restrict__ outf,
                          long outoff) {
  extern __shared__ __align__(16) char smem[];
  constexpr int ASTR = KDIM + 8;                 // bf16 elems per A row (16B-mult)
  __bf16* As = (__bf16*)smem;
  float*  Cs = (float*)(smem + 32 * ASTR * 2);   // 32 x NDIM f32

  const int  tid  = threadIdx.x;
  const long row0 = (long)blockIdx.x * 32;

  // ---------------- stage A tile (32 x KDIM) into LDS as bf16 ----------------
  if (AMODE == 0) {
    // 32 rows x 384 cols, gathered in float4 chunks (chunks never cross segs)
    for (int i = tid; i < 32 * 96; i += 256) {
      int r = i / 96;
      int c4 = (i - r * 96) * 4;
      long gr = row0 + r;
      int b = (int)(gr / N_EDGES);
      int e = (int)(gr - (long)b * N_EDGES);
      const float4* src;
      if (c4 < 128) {
        int t = EI[N_EDGES + e];  // tgt
        src = (const float4*)(X + ((long)b * N_NODES + t) * 128 + c4);
      } else if (c4 < 256) {
        src = (const float4*)(EF + ((long)b * N_EDGES + e) * 128 + (c4 - 128));
      } else {
        int s = EI[e];            // src
        src = (const float4*)(X + ((long)b * N_NODES + s) * 128 + (c4 - 256));
      }
      float4 v = *src;
      __bf16* dst = As + r * ASTR + c4;
      dst[0] = (__bf16)v.x; dst[1] = (__bf16)v.y;
      dst[2] = (__bf16)v.z; dst[3] = (__bf16)v.w;
    }
  } else if (AMODE == 1) {
#ifdef HAVE_TDM
    // Tensor Data Mover: one 2D-tile DMA (32 rows x KDIM bf16, row stride KDIM)
    // with LDS padding of 4 DWORDs after every KDIM/2 DWORDs to reproduce the
    // ASTR = KDIM+8 LDS row stride.  Dynamic LDS starts at offset 0.
    if (tid == 0) {
      unsigned long long ga = (unsigned long long)(const char*)(Abf + row0 * KDIM);
      u32x4 g0;
      g0[0] = 1u;                                        // count=1 (valid D#)
      g0[1] = 0u;                                        // lds_addr = 0
      g0[2] = (unsigned)(ga & 0xffffffffu);              // global_addr[31:0]
      g0[3] = (unsigned)((ga >> 32) & 0x1ffffffu) | (2u << 30);  // ga[56:32], type=2
      constexpr int padcode = (KDIM == 256) ? 6 : 5;     // 128 / 64 dwords per row
      i32x8 g1;
      g1[0] = (1 << 16) | (1 << 20) | (padcode << 22) | (3 << 25); // 2B, pad_en, +4dw
      g1[1] = (int)((unsigned)(KDIM & 0xffff) << 16);    // tensor_dim0[15:0]
      g1[2] = (int)(((unsigned)KDIM >> 16) |
                    ((unsigned)(160000 & 0xffff) << 16));// td0 hi | td1 lo
      g1[3] = (int)(((unsigned)160000 >> 16) |
                    ((unsigned)(KDIM & 0xffff) << 16));  // td1 hi | tile_dim0 = K
      g1[4] = 32;                                        // tile_dim1 = 32, tile_dim2 = 0
      g1[5] = KDIM;                                      // tensor_dim0_stride[31:0]
      g1[6] = 0;                                         // stride hi | td1_stride lo
      g1[7] = 0;
      i32x4 z4 = {0, 0, 0, 0};
      i32x8 z8 = {0, 0, 0, 0, 0, 0, 0, 0};
      __builtin_amdgcn_tensor_load_to_lds(g0, g1, z4, z4, z8, 0);
      __builtin_amdgcn_s_wait_tensorcnt((short)0);       // wave0 holds barrier below
    }
#else
    for (int i = tid; i < 32 * (KDIM / 8); i += 256) {
      int r = i / (KDIM / 8);
      int c = (i - r * (KDIM / 8)) * 8;
      bf16x8 v = *(const bf16x8*)(Abf + (row0 + r) * KDIM + c);
      *(bf16x8*)(As + r * ASTR + c) = v;
    }
#endif
  } else {
    for (int i = tid; i < 32 * (KDIM / 4); i += 256) {
      int r = i / (KDIM / 4);
      int c = (i - r * (KDIM / 4)) * 4;
      float4 v = *(const float4*)(Af32 + (row0 + r) * KDIM + c);
      __bf16* dst = As + r * ASTR + c;
      dst[0] = (__bf16)v.x; dst[1] = (__bf16)v.y;
      dst[2] = (__bf16)v.z; dst[3] = (__bf16)v.w;
    }
  }
  __syncthreads();

  // ---------------- WMMA main loop -------------------------------------------
  const int lane = tid & 31;
  const int wv   = tid >> 5;                 // 0..7
  const int mrow = wv >> 2;                  // 0..1 : which 16-row half
  const int ncb  = (wv & 3) * (NW * 16);     // this wave's N base
  const int m    = lane & 15;
  const int g    = lane >> 4;
  constexpr int KT = KDIM / 32;

  auto load_a = [&](int kt) -> v16bf {
    const __bf16* arow = As + (mrow * 16 + m) * ASTR + kt * 32;
    bf16x8 lo = *(const bf16x8*)(arow + g * 16);
    bf16x8 hi = *(const bf16x8*)(arow + g * 16 + 8);
    v16bf a;
#pragma unroll
    for (int i = 0; i < 8; ++i) { a[i] = lo[i]; a[8 + i] = hi[i]; }
    return a;
  };
  auto load_b = [&](int kt, int t) -> v16bf {
    const __bf16* wp = W + (size_t)(ncb + t * 16 + m) * KDIM + kt * 32 + g * 16;
    bf16x8 b0 = *(const bf16x8*)wp;
    bf16x8 b1 = *(const bf16x8*)(wp + 8);
    v16bf bb;
#pragma unroll
    for (int i = 0; i < 8; ++i) { bb[i] = b0[i]; bb[8 + i] = b1[i]; }
    return bb;
  };

  v8f acc[NW];
#pragma unroll
  for (int t = 0; t < NW; ++t)
#pragma unroll
    for (int r = 0; r < 8; ++r) acc[t][r] = 0.f;

  if (NW <= 4) {
    // one-deep software pipeline: fetch k-tile kt+1 while WMMAs consume kt
    v16bf a_cur = load_a(0);
    v16bf b_cur[NW];
#pragma unroll
    for (int t = 0; t < NW; ++t) b_cur[t] = load_b(0, t);
#pragma unroll
    for (int kt = 0; kt < KT; ++kt) {
      v16bf a_nxt = a_cur;
      v16bf b_nxt[NW];
      if (kt + 1 < KT) {
        a_nxt = load_a(kt + 1);
#pragma unroll
        for (int t = 0; t < NW; ++t) b_nxt[t] = load_b(kt + 1, t);
      } else {
#pragma unroll
        for (int t = 0; t < NW; ++t) b_nxt[t] = b_cur[t];
      }
#pragma unroll
      for (int t = 0; t < NW; ++t)
        acc[t] = __builtin_amdgcn_wmma_f32_16x16x32_bf16(
            false, a_cur, false, b_cur[t], (short)0, acc[t], false, false);
      a_cur = a_nxt;
#pragma unroll
      for (int t = 0; t < NW; ++t) b_cur[t] = b_nxt[t];
    }
  } else {
    // NW==10: too many VGPRs for double-buffering; batch loads per k-tile
    for (int kt = 0; kt < KT; ++kt) {
      v16bf a = load_a(kt);
      v16bf bb[NW];
#pragma unroll
      for (int t = 0; t < NW; ++t) bb[t] = load_b(kt, t);
#pragma unroll
      for (int t = 0; t < NW; ++t)
        acc[t] = __builtin_amdgcn_wmma_f32_16x16x32_bf16(
            false, a, false, bb[t], (short)0, acc[t], false, false);
    }
  }

  // ---------------- spill C tile to LDS --------------------------------------
#pragma unroll
  for (int t = 0; t < NW; ++t)
#pragma unroll
    for (int r = 0; r < 8; ++r)
      Cs[(mrow * 16 + r + 8 * g) * NDIM + ncb + t * 16 + m] = acc[t][r];
  __syncthreads();

  // ---------------- epilogue: one wave per row -------------------------------
  for (int rr = wv; rr < 32; rr += 8) {
    long gr = row0 + rr;
    float* crow = Cs + rr * NDIM;
    float mean = 0.f, rstd = 0.f;
    if (EMODE == 0 || EMODE == 2) {
      float s = 0.f, s2 = 0.f;
      for (int it = 0; it < NDIM / 32; ++it) {
        int c = lane + it * 32;
        float x = crow[c] + bias[c];
        s += x; s2 += x * x;
      }
      s = wave_sum(s); s2 = wave_sum(s2);
      mean = s / (float)NDIM;
      float var = s2 / (float)NDIM - mean * mean;
      rstd = rsqrtf(var + 1e-5f);
    }
    if (EMODE == 0) {
      for (int it = 0; it < NDIM / 32; ++it) {
        int c = lane + it * 32;
        float x = crow[c] + bias[c];
        float y = fmaxf((x - mean) * rstd * lng[c] + lnb[c], 0.f);
        outbf[gr * NDIM + c] = (__bf16)y;
      }
    } else if (EMODE == 1) {
      for (int it = 0; it < NDIM / 32; ++it) {
        int c = lane + it * 32;
        outbf[gr * NDIM + c] = (__bf16)(crow[c] + bias[c]);
      }
    } else if (EMODE == 2) {  // NDIM == 640
      int b = (int)(gr / N_EDGES);
      int e = (int)(gr - (long)b * N_EDGES);
      int tg = EI[N_EDGES + e];
      float* arow = agg + ((long)b * N_NODES + tg) * 256;
      auto lny = [&](int c) -> float {
        float x = crow[c] + bias[c];
        return fmaxf((x - mean) * rstd * lng[c] + lnb[c], 0.f);
      };
      for (int it = 0; it < 8; ++it) {            // msg = m[j] + m[384+j]
        int j = lane + it * 32;
        atomicAdd(arow + j, lny(j) + lny(j + 384));
      }
      for (int it = 0; it < 4; ++it) {            // new_e = m[256:384]
        int c = 256 + lane + it * 32;
        newe[gr * 128 + (c - 256)] = (__bf16)lny(c);
      }
    } else if (EMODE == 3) {
      for (int it = 0; it < NDIM / 32; ++it) {
        int c = lane + it * 32;
        float x = crow[c] + bias[c];
        outf[outoff + gr * NDIM + c] = 1.f / (1.f + expf(-x));
      }
    } else {  // EMODE == 4: H = (1 - sigmoid_z) * tanh_h, in place
      for (int it = 0; it < NDIM / 32; ++it) {
        int c = lane + it * 32;
        float x = crow[c] + bias[c];
        long oi = outoff + gr * NDIM + c;
        float s0 = outf[oi];
        outf[oi] = (1.f - s0) * tanhf(x);
      }
    }
  }
}

// f32 (Nout x Ksrc) -> bf16 (Nout x Kd) row-major (Kd <= Ksrc keeps W[:, :Kd])
__global__ void wprep(const float* __restrict__ W, __bf16* __restrict__ dst,
                      int N, int Kd, int Ks) {
  int i = blockIdx.x * 256 + threadIdx.x;
  if (i < N * Kd) {
    int n = i / Kd, k = i - n * Kd;
    dst[i] = (__bf16)W[(size_t)n * Ks + k];
  }
}

__global__ void zerok(float* p, int n) {
  int i = blockIdx.x * 256 + threadIdx.x;
  if (i < n) p[i] = 0.f;
}

extern "C" void kernel_launch(void* const* d_in, const int* in_sizes, int n_in,
                              void* d_out, int out_size, void* d_ws, size_t ws_size,
                              hipStream_t stream) {
  (void)in_sizes; (void)n_in; (void)out_size; (void)ws_size;
  const float* X  = (const float*)d_in[0];
  const int*   EI = (const int*)d_in[1];
  const float* EF = (const float*)d_in[2];
  const int GZ = 3, GH = 51;  // gate 'r' (base 27) is dead code

  // ---- workspace layout ----
  const size_t W_NN1L1 = 0,      W_NN1L2 = 98304,  W_NN2L1 = 262144,
               W_NN2L2 = 327680, W_NN3L1 = 393216, W_NN3L2 = 425984,
               W_EDGE  = 491520, W_NODE  = 557056, WG_ELEMS = 622592;
  char* ws = (char*)d_ws;
  __bf16* wz = (__bf16*)ws;
  __bf16* wh = wz + WG_ELEMS;
  size_t off = (2 * WG_ELEMS * 2 + 255) & ~(size_t)255;
  __bf16* bufA = (__bf16*)(ws + off); off += (size_t)160000 * 256 * 2;
  __bf16* bufB = (__bf16*)(ws + off); off += (size_t)160000 * 256 * 2;
  __bf16* newe = (__bf16*)(ws + off); off += (size_t)160000 * 128 * 2;
  float*  agg  = (float*)(ws + off);

  struct GP { int base; __bf16* w; };
  GP gps[2] = { {GZ, wz}, {GH, wh} };

  auto prep = [&](int pidx, __bf16* dst, int N, int Kd, int Ks) {
    int tot = N * Kd;
    wprep<<<(tot + 255) / 256, 256, 0, stream>>>((const float*)d_in[pidx], dst, N, Kd, Ks);
  };
  for (int gi = 0; gi < 2; ++gi) {
    GP gp = gps[gi];
    prep(gp.base + 0,  gp.w + W_NN1L1, 256, 384, 384);
    prep(gp.base + 4,  gp.w + W_NN1L2, 640, 256, 256);
    prep(gp.base + 8,  gp.w + W_NN2L1, 256, 256, 256);
    prep(gp.base + 12, gp.w + W_NN2L2, 256, 256, 256);
    prep(gp.base + 14, gp.w + W_NN3L1, 256, 128, 128);
    prep(gp.base + 18, gp.w + W_NN3L2, 256, 256, 256);
    prep(gp.base + 22, gp.w + W_EDGE,  256, 256, 512);  // edge.W[:, :256]
    prep(gp.base + 20, gp.w + W_NODE,  256, 256, 512);  // node.W[:, :256]
  }

  float* out = (float*)d_out;
  const long NODE_OUT = 0;
  const long EDGE_OUT = (long)N_BATCH * N_NODES * 256;  // 5,120,000

  const size_t L384 = 32 * (384 + 8) * 2 + 32 * 256 * 4;
  const size_t L256 = 32 * (256 + 8) * 2 + 32 * 256 * 4;
  const size_t L128 = 32 * (128 + 8) * 2 + 32 * 256 * 4;
  const size_t LMSG = 32 * (256 + 8) * 2 + 32 * 640 * 4;

  for (int gi = 0; gi < 2; ++gi) {   // gi=0: z (sigmoid), gi=1: h (tanh+combine)
    int base = gps[gi].base;
    __bf16* w = gps[gi].w;
    const float* p1b  = (const float*)d_in[base + 1];   // nn1.lin1.b
    const float* l1g  = (const float*)d_in[base + 2];
    const float* l1b  = (const float*)d_in[base + 3];
    const float* p2b  = (const float*)d_in[base + 5];   // nn1.lin2.b
    const float* l2g  = (const float*)d_in[base + 6];
    const float* l2b  = (const float*)d_in[base + 7];
    const float* n2ab = (const float*)d_in[base + 9];   // nn2.lin1.b
    const float* n2lg = (const float*)d_in[base + 10];
    const float* n2lb = (const float*)d_in[base + 11];
    const float* n2bb = (const float*)d_in[base + 13];  // nn2.lin2.b
    const float* n3ab = (const float*)d_in[base + 15];  // nn3.lin1.b
    const float* n3lg = (const float*)d_in[base + 16];
    const float* n3lb = (const float*)d_in[base + 17];
    const float* n3bb = (const float*)d_in[base + 19];  // nn3.lin2.b
    const float* ndb  = (const float*)d_in[base + 21];  // node.b
    const float* edb  = (const float*)d_in[base + 23];  // edge.b

    zerok<<<(20000 * 256 + 255) / 256, 256, 0, stream>>>(agg, 20000 * 256);

    // L1: gather-concat -> h1 (bufA)     [160000 x 384] x [384 x 256]
    tgcn_gemm<0, 0, 384, 256, 4><<<5000, 256, L384, stream>>>(
        X, EI, EF, nullptr, nullptr, w + W_NN1L1, p1b, l1g, l1b,
        bufA, nullptr, nullptr, nullptr, 0);
    // L2: h1 -> m (LN/ReLU) -> msg scatter + new_e     [x 256] x [256 x 640]
    tgcn_gemm<1, 2, 256, 640, 10><<<5000, 256, LMSG, stream>>>(
        X, EI, EF, bufA, nullptr, w + W_NN1L2, p2b, l2g, l2b,
        nullptr, agg, newe, nullptr, 0);
    // L3: new_e -> h3 (bufB)             [160000 x 128] x [128 x 256]
    tgcn_gemm<1, 0, 128, 256, 4><<<5000, 256, L128, stream>>>(
        X, EI, EF, newe, nullptr, w + W_NN3L1, n3ab, n3lg, n3lb,
        bufB, nullptr, nullptr, nullptr, 0);
    // L4: h3 -> gcn_e (bufA)
    tgcn_gemm<1, 1, 256, 256, 4><<<5000, 256, L256, stream>>>(
        X, EI, EF, bufB, nullptr, w + W_NN3L2, n3bb, nullptr, nullptr,
        bufA, nullptr, nullptr, nullptr, 0);
    // L5: gcn_e -> edge gate -> d_out edge region
    if (gi == 0)
      tgcn_gemm<1, 3, 256, 256, 4><<<5000, 256, L256, stream>>>(
          X, EI, EF, bufA, nullptr, w + W_EDGE, edb, nullptr, nullptr,
          nullptr, nullptr, nullptr, out, EDGE_OUT);
    else
      tgcn_gemm<1, 4, 256, 256, 4><<<5000, 256, L256, stream>>>(
          X, EI, EF, bufA, nullptr, w + W_EDGE, edb, nullptr, nullptr,
          nullptr, nullptr, nullptr, out, EDGE_OUT);
    // L6: agg -> h2 (bufB)               [20000 x 256] x [256 x 256]
    tgcn_gemm<2, 0, 256, 256, 4><<<625, 256, L256, stream>>>(
        X, EI, EF, nullptr, agg, w + W_NN2L1, n2ab, n2lg, n2lb,
        bufB, nullptr, nullptr, nullptr, 0);
    // L7: h2 -> gcn_x (bufA)
    tgcn_gemm<1, 1, 256, 256, 4><<<625, 256, L256, stream>>>(
        X, EI, EF, bufB, nullptr, w + W_NN2L2, n2bb, nullptr, nullptr,
        bufA, nullptr, nullptr, nullptr, 0);
    // L8: gcn_x -> node gate -> d_out node region
    if (gi == 0)
      tgcn_gemm<1, 3, 256, 256, 4><<<625, 256, L256, stream>>>(
          X, EI, EF, bufA, nullptr, w + W_NODE, ndb, nullptr, nullptr,
          nullptr, nullptr, nullptr, out, NODE_OUT);
    else
      tgcn_gemm<1, 4, 256, 256, 4><<<625, 256, L256, stream>>>(
          X, EI, EF, bufA, nullptr, w + W_NODE, ndb, nullptr, nullptr,
          nullptr, nullptr, nullptr, out, NODE_OUT);
  }
}
